// DeepFM_3066606649824
// MI455X (gfx1250) — matrix operand order repaired and checked by reading, not compiled, not run on hardware
//
#include <hip/hip_runtime.h>
#include <hip/hip_bf16.h>

// ---------------- problem constants (from reference) ----------------
constexpr int NF      = 26;
constexpr int VFIELD  = 8000;
constexpr int VTOT    = NF * VFIELD;        // 208000
constexpr int DEMB    = 16;
constexpr int DNN_IN  = NF * NF * DEMB;     // 10816 (== 169 * 64)
constexpr int H1      = 256;
constexpr int H2      = 128;
constexpr int BS      = 4096;
constexpr int NDENSE  = 13;
constexpr int NPAIR   = (NF * (NF - 1)) / 2; // 325

typedef _Float16 v16h __attribute__((ext_vector_type(16)));
typedef _Float16 v8h  __attribute__((ext_vector_type(8)));
typedef float    v8f  __attribute__((ext_vector_type(8)));

// ---------------- fp32 -> fp16 weight conversion ----------------
__global__ __launch_bounds__(256)
void f32_to_f16_kernel(const float* __restrict__ src, _Float16* __restrict__ dst, int n) {
    int i = blockIdx.x * 256 + threadIdx.x;
    if (i < n) dst[i] = (_Float16)src[i];
}

// ---------------- embedding gather + FM1/FM2 + dense side-input ----------------
// One block per sample. Builds dnn row (fp16) and fmsum[b] = fm1 + fm2 (+biases).
__global__ __launch_bounds__(256)
void embed_kernel(const int*   __restrict__ Xs,      // (BS, NF)
                  const float* __restrict__ Xd,      // (BS, 13)
                  const float* __restrict__ fm1e,    // (V, 1)
                  const float* __restrict__ bias0,   // (1,)
                  const float* __restrict__ fm1W,    // (1, 13)
                  const float* __restrict__ fm1b,    // (1,)
                  const float* __restrict__ emb,     // (NF, V, 16)
                  const float* __restrict__ dW,      // (DNN_IN, 13)
                  const float* __restrict__ db,      // (DNN_IN,)
                  _Float16*    __restrict__ dnn,     // (BS, DNN_IN) fp16
                  float*       __restrict__ fmsum)   // (BS,)
{
    __shared__ float At[DNN_IN];       // 43264 B
    __shared__ int   tok[NF];
    __shared__ float xd[NDENSE];
    __shared__ float red [256];
    __shared__ float red2[256];

    const int b   = blockIdx.x;
    const int tid = threadIdx.x;

    if (tid < NF) tok[tid] = Xs[b * NF + tid] + tid * VFIELD;
    if (tid >= 32 && tid < 32 + NDENSE) xd[tid - 32] = Xd[b * NDENSE + (tid - 32)];
    __syncthreads();

    // gather A and fuse relu(X_dense @ dense_W^T + dense_b)
    for (int e = tid; e < DNN_IN; e += 256) {
        const int d = e & 15;
        const int j = (e >> 4) % NF;
        const int i = e / (NF * DEMB);
        const float a = emb[((size_t)i * VTOT + tok[j]) * DEMB + d];
        At[e] = a;
        float s = db[e];
        #pragma unroll
        for (int k = 0; k < NDENSE; ++k) s += xd[k] * dW[e * NDENSE + k];
        s = s > 0.f ? s : 0.f;
        dnn[(size_t)b * DNN_IN + e] = (_Float16)(a + s);
    }
    __syncthreads();

    // fm2 = sum_{i<j} sum_d A[i,j,d] * A[j,i,d]
    float acc = 0.f;
    for (int t = tid; t < NPAIR * DEMB; t += 256) {
        const int d = t & 15;
        int pp = t >> 4;
        int i = 0;
        while (pp >= NF - 1 - i) { pp -= NF - 1 - i; ++i; }
        const int j = i + 1 + pp;
        acc += At[(i * NF + j) * DEMB + d] * At[(j * NF + i) * DEMB + d];
    }
    // fm1 partial terms
    float acc1 = 0.f;
    if (tid < NF) acc1 += fm1e[tok[tid]];
    if (tid >= 32 && tid < 32 + NDENSE) acc1 += xd[tid - 32] * fm1W[tid - 32];

    red[tid] = acc; red2[tid] = acc1;
    __syncthreads();
    for (int s = 128; s > 0; s >>= 1) {
        if (tid < s) { red[tid] += red[tid + s]; red2[tid] += red2[tid + s]; }
        __syncthreads();
    }
    if (tid == 0) fmsum[b] = red[0] + red2[0] + bias0[0] + fm1b[0];
}

// ---------------- WMMA GEMM fragments ----------------
struct AFrag { union { v16h v; v8h h[2]; }; };
struct BFrag { v16h b[4]; };

__device__ __forceinline__
void load_a(const _Float16* __restrict__ aRow, int kb, int aSub0, int aSub1, AFrag& a) {
    a.h[0] = *reinterpret_cast<const v8h*>(aRow + kb + aSub0);
    a.h[1] = *reinterpret_cast<const v8h*>(aRow + kb + aSub1);
}

__device__ __forceinline__
void load_b(const _Float16* __restrict__ bR0, const _Float16* __restrict__ bR1,
            const _Float16* __restrict__ bR2, const _Float16* __restrict__ bR3,
            int kb, BFrag& f) {
    f.b[0] = *reinterpret_cast<const v16h*>(bR0 + kb);
    f.b[1] = *reinterpret_cast<const v16h*>(bR1 + kb);
    f.b[2] = *reinterpret_cast<const v16h*>(bR2 + kb);
    f.b[3] = *reinterpret_cast<const v16h*>(bR3 + kb);
}

__device__ __forceinline__
void mma4(const AFrag& a, const BFrag& f, v8f& c0, v8f& c1, v8f& c2, v8f& c3) {
    c0 = __builtin_amdgcn_wmma_f32_16x16x32_f16(false, a.v, false, f.b[0], (short)0, c0, false, false);
    c1 = __builtin_amdgcn_wmma_f32_16x16x32_f16(false, a.v, false, f.b[1], (short)0, c1, false, false);
    c2 = __builtin_amdgcn_wmma_f32_16x16x32_f16(false, a.v, false, f.b[2], (short)0, c2, false, false);
    c3 = __builtin_amdgcn_wmma_f32_16x16x32_f16(false, a.v, false, f.b[3], (short)0, c3, false, false);
}

// ---------------- WMMA GEMM: C(MxN) = A(MxK,f16) * B(NxK,f16)^T + bias ----------------
// 256 threads = 8 waves (4 in M x 2 in N); block tile 64 x 128.
// Each wave computes 16 rows x 64 cols = 4 wmma tiles per K-step.
// Register ping-pong double-buffering: loads for K-step k+32 are issued before the
// WMMAs of K-step k, so the compiler emits partial s_wait_loadcnt (latency hiding)
// instead of a full drain before every matrix op.  Requires K % 64 == 0.
// Fragment k order: contiguous k = 16*halfSel + s in both A and B (wave32, 16-bit A 16x32 / B 32x16).
__global__ __launch_bounds__(256)
void wmma_gemm_kernel(const _Float16* __restrict__ A,    // M x K row-major
                      const _Float16* __restrict__ B,    // N x K row-major
                      const float*    __restrict__ bias, // N
                      float*          __restrict__ C,    // M x N row-major
                      int K, int N)
{
    const int lane    = threadIdx.x & 31;
    const int wave    = threadIdx.x >> 5;
    const int waveM   = wave & 3;      // 0..3 -> 16-row strips within 64-row block tile
    const int waveN   = wave >> 2;     // 0..1 -> 64-col strips within 128-col block tile
    const int halfSel = lane >> 4;     // 0: lanes 0-15, 1: lanes 16-31
    const int l16     = lane & 15;

    const int mBase = blockIdx.x * 64 + waveM * 16;
    const int nBase = blockIdx.y * 128 + waveN * 64;

    // A row for this lane: M = mBase + l16; K halves split by lane group.
    const _Float16* aRow  = A + (size_t)(mBase + l16) * K;
    const int aSub0 = halfSel * 16;       // K = kb + aSub0 + 0..7
    const int aSub1 = halfSel * 16 + 8;   // K = kb + aSub1 + 0..7

    // B rows: N = nBase + t*16 + l16; 16 contiguous K per lane.
    const int bSub = halfSel * 16;
    const _Float16* bR0 = B + (size_t)(nBase +  0 + l16) * K + bSub;
    const _Float16* bR1 = B + (size_t)(nBase + 16 + l16) * K + bSub;
    const _Float16* bR2 = B + (size_t)(nBase + 32 + l16) * K + bSub;
    const _Float16* bR3 = B + (size_t)(nBase + 48 + l16) * K + bSub;

    v8f c0 = {}, c1 = {}, c2 = {}, c3 = {};

    AFrag aP, aQ;
    BFrag bP, bQ;

    // prologue: stage K-step 0
    load_a(aRow, 0, aSub0, aSub1, aP);
    load_b(bR0, bR1, bR2, bR3, 0, bP);

    int kb = 0;
    for (; kb < K - 64; kb += 64) {
        __builtin_prefetch(aRow + kb + 128, 0, 1);      // global_prefetch_b8
        // stage k+32, then compute k
        load_a(aRow, kb + 32, aSub0, aSub1, aQ);
        load_b(bR0, bR1, bR2, bR3, kb + 32, bQ);
        mma4(aP, bP, c0, c1, c2, c3);
        // stage k+64, then compute k+32
        load_a(aRow, kb + 64, aSub0, aSub1, aP);
        load_b(bR0, bR1, bR2, bR3, kb + 64, bP);
        mma4(aQ, bQ, c0, c1, c2, c3);
    }
    // epilogue: last 64 of K (kb == K-64 here)
    load_a(aRow, kb + 32, aSub0, aSub1, aQ);
    load_b(bR0, bR1, bR2, bR3, kb + 32, bQ);
    mma4(aP, bP, c0, c1, c2, c3);
    mma4(aQ, bQ, c0, c1, c2, c3);

    // C/D layout: VGPR v -> M = mBase + halfSel*8 + v ; N = nBase + t*16 + l16
    const int row0 = mBase + halfSel * 8;
    const int col  = nBase + l16;
    #pragma unroll
    for (int v = 0; v < 8; ++v) {
        float* cr = C + (size_t)(row0 + v) * N;
        cr[col     ] = c0[v] + bias[col     ];
        cr[col + 16] = c1[v] + bias[col + 16];
        cr[col + 32] = c2[v] + bias[col + 32];
        cr[col + 48] = c3[v] + bias[col + 48];
    }
}

// ---------------- batch-norm statistics: one block per column ----------------
__global__ __launch_bounds__(256)
void bn_stats_kernel(const float* __restrict__ Y, int rows, int cols,
                     float* __restrict__ mean, float* __restrict__ rstd)
{
    const int c = blockIdx.x, tid = threadIdx.x;
    float s = 0.f, s2 = 0.f;
    for (int r = tid; r < rows; r += 256) {
        const float v = Y[(size_t)r * cols + c];
        s += v; s2 += v * v;
    }
    __shared__ float rs[256], rq[256];
    rs[tid] = s; rq[tid] = s2;
    __syncthreads();
    for (int st = 128; st > 0; st >>= 1) {
        if (tid < st) { rs[tid] += rs[tid + st]; rq[tid] += rq[tid + st]; }
        __syncthreads();
    }
    if (tid == 0) {
        const float m   = rs[0] / rows;
        const float var = rq[0] / rows - m * m;   // population variance (jnp.var)
        mean[c] = m;
        rstd[c] = rsqrtf(var + 1e-5f);
    }
}

// ---------------- BN + ReLU -> fp16 activations ----------------
__global__ __launch_bounds__(256)
void bn_relu_h_kernel(const float* __restrict__ Y,
                      const float* __restrict__ mean, const float* __restrict__ rstd,
                      const float* __restrict__ g,    const float* __restrict__ beta,
                      _Float16* __restrict__ out, int colMask, int total)
{
    const int idx = blockIdx.x * 256 + threadIdx.x;
    if (idx >= total) return;
    const int c = idx & colMask;   // cols is a power of two
    float v = (Y[idx] - mean[c]) * rstd[c] * g[c] + beta[c];
    out[idx] = (_Float16)(v > 0.f ? v : 0.f);
}

// ---------------- final: BN2 + ReLU + dot(Wout) + sigmoid(fm + out) ----------------
__global__ __launch_bounds__(128)
void final_kernel(const float* __restrict__ Y2,
                  const float* __restrict__ mean, const float* __restrict__ rstd,
                  const float* __restrict__ g,    const float* __restrict__ beta,
                  const float* __restrict__ Wout, const float* __restrict__ bout,
                  const float* __restrict__ fmsum, float* __restrict__ out)
{
    const int b = blockIdx.x, tid = threadIdx.x;
    float v = (Y2[(size_t)b * H2 + tid] - mean[tid]) * rstd[tid] * g[tid] + beta[tid];
    v = v > 0.f ? v : 0.f;
    v *= Wout[tid];
    __shared__ float rs[128];
    rs[tid] = v;
    __syncthreads();
    for (int st = 64; st > 0; st >>= 1) {
        if (tid < st) rs[tid] += rs[tid + st];
        __syncthreads();
    }
    if (tid == 0) {
        const float z = fmsum[b] + rs[0] + bout[0];
        out[b] = 1.f / (1.f + __expf(-z));
    }
}

// ---------------- host-side orchestration ----------------
extern "C" void kernel_launch(void* const* d_in, const int* in_sizes, int n_in,
                              void* d_out, int out_size, void* d_ws, size_t ws_size,
                              hipStream_t stream) {
    (void)in_sizes; (void)n_in; (void)out_size; (void)ws_size;

    const int*   Xs    = (const int*)  d_in[0];
    const float* Xd    = (const float*)d_in[1];
    const float* fm1e  = (const float*)d_in[2];
    const float* bias0 = (const float*)d_in[3];
    const float* fm1W  = (const float*)d_in[4];
    const float* fm1b  = (const float*)d_in[5];
    const float* emb   = (const float*)d_in[6];
    const float* dW    = (const float*)d_in[7];
    const float* db    = (const float*)d_in[8];
    const float* W1    = (const float*)d_in[9];
    const float* b1    = (const float*)d_in[10];
    const float* g1    = (const float*)d_in[11];
    const float* be1   = (const float*)d_in[12];
    const float* W2    = (const float*)d_in[13];
    const float* b2    = (const float*)d_in[14];
    const float* g2    = (const float*)d_in[15];
    const float* be2   = (const float*)d_in[16];
    const float* Wout  = (const float*)d_in[17];
    const float* bout  = (const float*)d_in[18];
    float* out = (float*)d_out;

    // workspace carve-up (256B aligned)
    char* ws = (char*)d_ws;
    size_t off = 0;
    auto carve = [&](size_t bytes) -> char* {
        char* p = ws + off;
        off += (bytes + 255) & ~size_t(255);
        return p;
    };
    _Float16* dnn   = (_Float16*)carve((size_t)BS * DNN_IN * sizeof(_Float16)); // 88.6 MB
    _Float16* W1h   = (_Float16*)carve((size_t)H1 * DNN_IN * sizeof(_Float16)); //  5.5 MB
    _Float16* W2h   = (_Float16*)carve((size_t)H2 * H1     * sizeof(_Float16));
    float*    Y1    = (float*)   carve((size_t)BS * H1 * sizeof(float));
    _Float16* H1act = (_Float16*)carve((size_t)BS * H1 * sizeof(_Float16));
    float*    Y2    = (float*)   carve((size_t)BS * H2 * sizeof(float));
    float*    fmsum = (float*)   carve((size_t)BS * sizeof(float));
    float*    m1    = (float*)   carve(H1 * sizeof(float));
    float*    r1    = (float*)   carve(H1 * sizeof(float));
    float*    m2    = (float*)   carve(H2 * sizeof(float));
    float*    r2    = (float*)   carve(H2 * sizeof(float));

    // 1) weight precision conversion (fp32 -> fp16 for WMMA)
    f32_to_f16_kernel<<<(H1 * DNN_IN + 255) / 256, 256, 0, stream>>>(W1, W1h, H1 * DNN_IN);
    f32_to_f16_kernel<<<(H2 * H1     + 255) / 256, 256, 0, stream>>>(W2, W2h, H2 * H1);

    // 2) embedding gather + FM terms + dense side-input (writes dnn fp16, fmsum)
    embed_kernel<<<BS, 256, 0, stream>>>(Xs, Xd, fm1e, bias0, fm1W, fm1b, emb, dW, db, dnn, fmsum);

    // 3) GEMM1: Y1 = dnn @ W1^T + b1   (4096 x 10816 x 256, WMMA f16)
    //    block tile 64x128 -> dnn streamed only 2x from HBM; W1h L2-resident.
    wmma_gemm_kernel<<<dim3(BS / 64, H1 / 128), 256, 0, stream>>>(dnn, W1h, b1, Y1, DNN_IN, H1);

    // 4) BN1 stats + apply (+ReLU) -> fp16
    bn_stats_kernel<<<H1, 256, 0, stream>>>(Y1, BS, H1, m1, r1);
    bn_relu_h_kernel<<<(BS * H1) / 256, 256, 0, stream>>>(Y1, m1, r1, g1, be1, H1act, H1 - 1, BS * H1);

    // 5) GEMM2: Y2 = H1act @ W2^T + b2  (4096 x 256 x 128, WMMA f16)
    wmma_gemm_kernel<<<dim3(BS / 64, H2 / 128), 256, 0, stream>>>(H1act, W2h, b2, Y2, H1, H2);

    // 6) BN2 stats + fused BN/ReLU/dot/sigmoid
    bn_stats_kernel<<<H2, 256, 0, stream>>>(Y2, BS, H2, m2, r2);
    final_kernel<<<BS, 128, 0, stream>>>(Y2, m2, r2, g2, be2, Wout, bout, fmsum, out);
}
